// EGL_13709535608834
// MI455X (gfx1250) — matrix operand order repaired and checked by reading, not compiled, not run on hardware
//
#include <hip/hip_runtime.h>
#include <hip/hip_bf16.h>

// ---------------------------------------------------------------------------
// MI455X (gfx1250) implementation: cosine-graph SAGE + all-pairs combiner.
// All GEMMs via v_wmma_f32_16x16x32_f16 (f16 in, f32 accumulate).
// ---------------------------------------------------------------------------

#define NN   2048
#define DD   64
#define PP   32
#define NSN  512
#define KTOT 4128          // 32 (pref) + 2048 (es) + 2048 (dist)
#define INLEN 4131
#define LEAKY 0.01f
#define COSEPS 1e-8f

typedef __attribute__((ext_vector_type(16))) _Float16 v16h;
typedef __attribute__((ext_vector_type(8)))  _Float16 v8h;
typedef __attribute__((ext_vector_type(8)))  float    v8f;

// ---- workspace layout (bytes, 256-aligned) --------------------------------
#define OFF_A     ((size_t)0)                         // A half  [2048 x 4128]
#define OFF_B     (OFF_A + (size_t)NN*KTOT*2)         // B half  [2048 x 4128] (adj aliases here first)
#define OFF_XN    (OFF_B + (size_t)NN*KTOT*2)         // xn half [2048 x 64]
#define OFF_EMBT  (OFF_XN + (size_t)NN*DD*2)          // emb^T half [64 x 2048]
#define OFF_AGG   (OFF_EMBT + (size_t)NN*DD*2)        // agg f32 [2048 x 64]
#define OFF_DEG   (OFF_AGG + (size_t)NN*DD*4)
#define OFF_KEEP  (OFF_DEG + (size_t)NN*4)
#define OFF_AV    (OFF_KEEP + (size_t)NN*4)
#define OFF_BV    (OFF_AV + (size_t)NN*4)
#define OFF_C0    (OFF_BV + (size_t)NN*4)
#define OFF_C1    (OFF_C0 + (size_t)NN*4)

// ---- WMMA helpers ---------------------------------------------------------
__device__ __forceinline__ v8f wmma32(v16h a, v16h b, v8f c) {
  return __builtin_amdgcn_wmma_f32_16x16x32_f16(false, a, false, b, (short)0, c,
                                                false, false);
}

__device__ __forceinline__ v16h pack2(const _Float16* p0, const _Float16* p1) {
  v8h lo = *(const v8h*)p0;
  v8h hi = *(const v8h*)p1;
  v16h r;
#pragma unroll
  for (int e = 0; e < 8; ++e) { r[e] = lo[e]; r[8 + e] = hi[e]; }
  return r;
}

// A fragment: 16x32 f16, rows row0..row0+15, K slab [k0, k0+32). ISA layout:
// lane L holds M=L&15; elems 0..7 -> K = (L>>4)*8 + 0..7, elems 8..15 -> +16.
__device__ __forceinline__ v16h loadA(const _Float16* base, int ld, int row0,
                                      int k0, int lane) {
  const _Float16* p =
      base + (size_t)(row0 + (lane & 15)) * ld + k0 + ((lane >> 4) << 3);
  return pack2(p, p + 16);
}

// B fragment from row-major B^T storage (row j of storage == column j of B):
// lane L holds N=L&15; elems 0..7 -> K = k0 + (L>>4)*8 + 0..7, elems 8..15 -> +16.
__device__ __forceinline__ v16h loadBT(const _Float16* base, int ld, int col0,
                                       int k0, int lane) {
  const _Float16* p =
      base + (size_t)(col0 + (lane & 15)) * ld + k0 + ((lane >> 4) << 3);
  return pack2(p, p + 16);
}

__device__ __forceinline__ float wave_sum(float v) {
#pragma unroll
  for (int o = 16; o > 0; o >>= 1) v += __shfl_xor(v, o, 32);
  return v;
}

// ---- K0: row norms -> xn half + emb^T half --------------------------------
__global__ __launch_bounds__(256) void k_norm(const float* __restrict__ emb,
                                              _Float16* __restrict__ xn,
                                              _Float16* __restrict__ embT) {
  int lane = threadIdx.x & 31;
  int row = blockIdx.x * 8 + (threadIdx.x >> 5);
  float e0 = emb[(size_t)row * DD + lane];
  float e1 = emb[(size_t)row * DD + 32 + lane];
  float ss = wave_sum(e0 * e0 + e1 * e1);
  float inv = 1.0f / fmaxf(sqrtf(ss), COSEPS);
  xn[(size_t)row * DD + lane]      = (_Float16)(e0 * inv);
  xn[(size_t)row * DD + 32 + lane] = (_Float16)(e1 * inv);
  embT[(size_t)lane * NN + row]        = (_Float16)e0;
  embT[(size_t)(32 + lane) * NN + row] = (_Float16)e1;
}

// ---- K1a/K1b: keep mask ---------------------------------------------------
__global__ void k_keep_zero(float* __restrict__ keep) {
  int i = blockIdx.x * blockDim.x + threadIdx.x;
  if (i < NN) keep[i] = 0.0f;
}
__global__ void k_keep_set(const int* __restrict__ stops,
                           float* __restrict__ keep) {
  int i = blockIdx.x * blockDim.x + threadIdx.x;
  if (i < NSN) keep[stops[i]] = 1.0f;
}

// ---- K2: sim = xn @ xn^T, threshold -> adj (f16 0/1), branchless ----------
__global__ __launch_bounds__(256) void k_sim_adj(const _Float16* __restrict__ xn,
                                                 const float* __restrict__ keep,
                                                 _Float16* __restrict__ adj) {
  int lane = threadIdx.x & 31;
  int tile = blockIdx.x * 8 + (threadIdx.x >> 5);  // 0..16383
  int ti = tile >> 7, tj = tile & 127;
  v8f c = {};
  c = wmma32(loadA(xn, DD, ti * 16, 0, lane),  loadBT(xn, DD, tj * 16, 0, lane),  c);
  c = wmma32(loadA(xn, DD, ti * 16, 32, lane), loadBT(xn, DD, tj * 16, 32, lane), c);
  int n = tj * 16 + (lane & 15);
  float kn = keep[n];
  int mb = ti * 16 + ((lane >> 4) << 3);
#pragma unroll
  for (int r = 0; r < 8; ++r) {
    int m = mb + r;
    float km = keep[m];                         // unconditional load
    bool pass = (c[r] > 0.5f) & (m != n);       // no short-circuit -> cndmask
    float v = pass ? km * kn : 0.0f;            // keep values are exact 0/1
    adj[(size_t)m * NN + n] = (_Float16)v;
  }
}

// ---- K3: deg = max(rowsum(adj), 1) ----------------------------------------
__global__ __launch_bounds__(256) void k_deg(const _Float16* __restrict__ adj,
                                             float* __restrict__ deg) {
  __shared__ float red[256];
  int i = blockIdx.x;
  float s = 0.0f;
  for (int j = threadIdx.x; j < NN; j += 256) s += (float)adj[(size_t)i * NN + j];
  red[threadIdx.x] = s; __syncthreads();
  for (int o = 128; o > 0; o >>= 1) {
    if (threadIdx.x < o) red[threadIdx.x] += red[threadIdx.x + o];
    __syncthreads();
  }
  if (threadIdx.x == 0) deg[i] = fmaxf(red[0], 1.0f);
}

// ---- K4: agg = adj @ emb (un-normalized); B from emb^T --------------------
__global__ __launch_bounds__(256) void k_agg(const _Float16* __restrict__ adj,
                                             const _Float16* __restrict__ embT,
                                             float* __restrict__ agg) {
  int lane = threadIdx.x & 31;
  int tile = blockIdx.x * 8 + (threadIdx.x >> 5);  // 0..511 : 128 M x 4 N
  int ti = tile >> 2, tj = tile & 3;
  v8f c = {};
  for (int k0 = 0; k0 < NN; k0 += 32)
    c = wmma32(loadA(adj, NN, ti * 16, k0, lane),
               loadBT(embT, NN, tj * 16, k0, lane), c);
  int n = tj * 16 + (lane & 15);
#pragma unroll
  for (int r = 0; r < 8; ++r) {
    int m = ti * 16 + r + ((lane >> 4) << 3);
    agg[(size_t)m * DD + n] = c[r];
  }
}

// ---- K5: pref (f32 -> f16 into A[:,0:32]), a[i], b[i] ---------------------
__global__ __launch_bounds__(256) void k_pref(
    const float* __restrict__ agg, const float* __restrict__ deg,
    const float* __restrict__ emb, const float* __restrict__ lin_l_w,
    const float* __restrict__ lin_l_b, const float* __restrict__ lin_r_w,
    const float* __restrict__ edge_w, _Float16* __restrict__ Ah,
    float* __restrict__ av, float* __restrict__ bv) {
  int p = threadIdx.x & 31;
  int i = blockIdx.x * 8 + (threadIdx.x >> 5);
  float inv = 1.0f / deg[i];
  float acc = lin_l_b[p];
#pragma unroll 8
  for (int d = 0; d < DD; ++d) {
    acc += agg[(size_t)i * DD + d] * inv * lin_l_w[p * DD + d];
    acc += emb[(size_t)i * DD + d] * lin_r_w[p * DD + d];
  }
  Ah[(size_t)i * KTOT + p] = (_Float16)acc;
  float sa = wave_sum(acc * edge_w[p]);
  float sb = wave_sum(acc * edge_w[PP + p]);
  if (p == 0) { av[i] = sa; bv[i] = sb; }
}

// ---- K6: fill es + dist halves of A ---------------------------------------
__global__ __launch_bounds__(256) void k_esdist(const float* __restrict__ av,
                                                const float* __restrict__ bv,
                                                const float* __restrict__ edge_b,
                                                const float* __restrict__ dist,
                                                _Float16* __restrict__ Ah) {
  size_t idx = (size_t)blockIdx.x * 256 + threadIdx.x;
  int i = (int)(idx >> 11), j = (int)(idx & 2047);
  float e = av[i] + bv[j] + edge_b[0];
  e = (e >= 0.0f) ? e : LEAKY * e;
  Ah[(size_t)i * KTOT + PP + j]        = (_Float16)e;
  Ah[(size_t)i * KTOT + PP + NN + j]   = (_Float16)dist[idx];
}

// ---- K7a: B half = comb_w[:, :4128] ---------------------------------------
__global__ __launch_bounds__(256) void k_bbuild(const float* __restrict__ comb_w,
                                                _Float16* __restrict__ Bh) {
  size_t idx = (size_t)blockIdx.x * 256 + threadIdx.x;
  if (idx >= (size_t)NN * KTOT) return;
  int j = (int)(idx / KTOT), k = (int)(idx % KTOT);
  Bh[idx] = (_Float16)comb_w[(size_t)j * INLEN + k];
}

// ---- K7b: epilogue constants ----------------------------------------------
__global__ void k_c01(const float* __restrict__ comb_w,
                      const float* __restrict__ comb_b,
                      const int* __restrict__ weekday,
                      const int* __restrict__ vehicles,
                      float* __restrict__ c0, float* __restrict__ c1) {
  int j = blockIdx.x * blockDim.x + threadIdx.x;
  if (j >= NN) return;
  float wd = (float)weekday[0], vh = (float)vehicles[0];
  const float* w = comb_w + (size_t)j * INLEN;
  c0[j] = wd * w[KTOT] + vh * w[KTOT + 1] + comb_b[j];
  c1[j] = w[KTOT + 2];
}

// ---- K8: main GEMM: out = A @ B^T + c0 + keep*c1 --------------------------
// 8 waves/block; each wave owns a 32x64 patch (2 M-tiles x 4 N-tiles),
// block tile 128x128. 6 fragment loads feed 8 WMMAs per K-step.
__global__ __launch_bounds__(256) void k_biggemm(const _Float16* __restrict__ A,
                                                 const _Float16* __restrict__ B,
                                                 const float* __restrict__ c0,
                                                 const float* __restrict__ c1,
                                                 const float* __restrict__ keep,
                                                 float* __restrict__ out) {
  int lane = threadIdx.x & 31;
  int wave = threadIdx.x >> 5;
  int m0 = blockIdx.y * 128 + (wave >> 1) * 32;  // 4 wave-rows
  int n0 = blockIdx.x * 128 + (wave & 1) * 64;   // 2 wave-cols
  v8f acc[2][4] = {};
  for (int k0 = 0; k0 < KTOT; k0 += 32) {
    v16h a0 = loadA(A, KTOT, m0, k0, lane);
    v16h a1 = loadA(A, KTOT, m0 + 16, k0, lane);
    v16h b0 = loadBT(B, KTOT, n0, k0, lane);
    v16h b1 = loadBT(B, KTOT, n0 + 16, k0, lane);
    v16h b2 = loadBT(B, KTOT, n0 + 32, k0, lane);
    v16h b3 = loadBT(B, KTOT, n0 + 48, k0, lane);
    acc[0][0] = wmma32(a0, b0, acc[0][0]);
    acc[0][1] = wmma32(a0, b1, acc[0][1]);
    acc[0][2] = wmma32(a0, b2, acc[0][2]);
    acc[0][3] = wmma32(a0, b3, acc[0][3]);
    acc[1][0] = wmma32(a1, b0, acc[1][0]);
    acc[1][1] = wmma32(a1, b1, acc[1][1]);
    acc[1][2] = wmma32(a1, b2, acc[1][2]);
    acc[1][3] = wmma32(a1, b3, acc[1][3]);
  }
  int nc = lane & 15;
  int mrb = (lane >> 4) << 3;
#pragma unroll
  for (int mi = 0; mi < 2; ++mi) {
#pragma unroll
    for (int r = 0; r < 8; ++r) {
      int m = m0 + mi * 16 + r + mrb;
      float km = keep[m];
#pragma unroll
      for (int ni = 0; ni < 4; ++ni) {
        int n = n0 + ni * 16 + nc;
        out[(size_t)m * NN + n] = acc[mi][ni][r] + c0[n] + km * c1[n];
      }
    }
  }
}

// ---- K9: row log-softmax ---------------------------------------------------
__global__ __launch_bounds__(256) void k_lsm(float* __restrict__ out) {
  __shared__ float red[256];
  float* row = out + (size_t)blockIdx.x * NN;
  float m = -INFINITY;
  for (int j = threadIdx.x; j < NN; j += 256) m = fmaxf(m, row[j]);
  red[threadIdx.x] = m; __syncthreads();
  for (int o = 128; o > 0; o >>= 1) {
    if (threadIdx.x < o) red[threadIdx.x] = fmaxf(red[threadIdx.x], red[threadIdx.x + o]);
    __syncthreads();
  }
  m = red[0]; __syncthreads();
  float s = 0.0f;
  for (int j = threadIdx.x; j < NN; j += 256) s += expf(row[j] - m);
  red[threadIdx.x] = s; __syncthreads();
  for (int o = 128; o > 0; o >>= 1) {
    if (threadIdx.x < o) red[threadIdx.x] += red[threadIdx.x + o];
    __syncthreads();
  }
  float lse = m + logf(red[0]);
  __syncthreads();
  for (int j = threadIdx.x; j < NN; j += 256) row[j] = row[j] - lse;
}

// ---------------------------------------------------------------------------
extern "C" void kernel_launch(void* const* d_in, const int* in_sizes, int n_in,
                              void* d_out, int out_size, void* d_ws,
                              size_t ws_size, hipStream_t stream) {
  const float* dist     = (const float*)d_in[0];
  const float* emb      = (const float*)d_in[1];
  const float* lin_l_w  = (const float*)d_in[2];
  const float* lin_l_b  = (const float*)d_in[3];
  const float* lin_r_w  = (const float*)d_in[4];
  const float* edge_w   = (const float*)d_in[5];
  const float* edge_b   = (const float*)d_in[6];
  const float* comb_w   = (const float*)d_in[7];
  const float* comb_b   = (const float*)d_in[8];
  const int*   stops    = (const int*)d_in[9];
  const int*   weekday  = (const int*)d_in[10];
  const int*   vehicles = (const int*)d_in[11];
  float* out = (float*)d_out;

  char* ws = (char*)d_ws;
  _Float16* Ah   = (_Float16*)(ws + OFF_A);
  _Float16* Bh   = (_Float16*)(ws + OFF_B);
  _Float16* adj  = (_Float16*)(ws + OFF_B);   // alias: consumed before Bh built
  _Float16* xn   = (_Float16*)(ws + OFF_XN);
  _Float16* embT = (_Float16*)(ws + OFF_EMBT);
  float* agg  = (float*)(ws + OFF_AGG);
  float* deg  = (float*)(ws + OFF_DEG);
  float* keep = (float*)(ws + OFF_KEEP);
  float* av   = (float*)(ws + OFF_AV);
  float* bv   = (float*)(ws + OFF_BV);
  float* c0   = (float*)(ws + OFF_C0);
  float* c1   = (float*)(ws + OFF_C1);

  k_norm<<<NN / 8, 256, 0, stream>>>(emb, xn, embT);
  k_keep_zero<<<NN / 256, 256, 0, stream>>>(keep);
  k_keep_set<<<NSN / 256, 256, 0, stream>>>(stops, keep);
  k_sim_adj<<<(128 * 128) / 8, 256, 0, stream>>>(xn, keep, adj);
  k_deg<<<NN, 256, 0, stream>>>(adj, deg);
  k_agg<<<512 / 8, 256, 0, stream>>>(adj, embT, agg);
  k_pref<<<NN / 8, 256, 0, stream>>>(agg, deg, emb, lin_l_w, lin_l_b, lin_r_w,
                                     edge_w, Ah, av, bv);
  k_esdist<<<(NN * NN) / 256, 256, 0, stream>>>(av, bv, edge_b, dist, Ah);
  k_bbuild<<<(int)(((size_t)NN * KTOT + 255) / 256), 256, 0, stream>>>(comb_w, Bh);
  k_c01<<<NN / 256, 256, 0, stream>>>(comb_w, comb_b, weekday, vehicles, c0, c1);
  dim3 grid(NN / 128, NN / 128);
  k_biggemm<<<grid, 256, 0, stream>>>(Ah, Bh, c0, c1, keep, out);
  k_lsm<<<NN, 256, 0, stream>>>(out);
}
